// SampledGAT_15590731284987
// MI455X (gfx1250) — compile-verified, matrix operand order checked by reading, not run on hardware
//
#include <hip/hip_runtime.h>
#include <hip/hip_bf16.h>

// ---------------------------------------------------------------------------
// Sampled 2-layer GAT for MI455X (gfx1250, wave32, WMMA f16->f32).
//
// Compute-bound (~105 GFLOP vs ~175 MB HBM traffic) -> all GEMMs through
// v_wmma_f32_16x16x32_f16, attention fused with online softmax so K/V never
// touch memory.
//   block = 256 threads = 8 waves; block owns 16 nodes (M=16);
//   wave w owns head w = output columns [w*64, w*64+64) = 4 WMMA N-tiles.
// Q is parked in LDS after its GEMM to cut VGPR pressure in the fanout loop.
// Layer-2 neighbor staging uses the Tensor Data Mover (tensor_load_to_lds)
// with D#-descriptor LDS padding; completion via s_wait_tensorcnt.
// ---------------------------------------------------------------------------

typedef __attribute__((ext_vector_type(8)))  _Float16 v8h;
typedef __attribute__((ext_vector_type(16))) _Float16 v16h;
typedef __attribute__((ext_vector_type(8)))  float    v8f;
typedef __attribute__((ext_vector_type(4)))  unsigned u32x4;
typedef __attribute__((ext_vector_type(8)))  unsigned u32x8;

#define NUM_NODES 500000
#define EMB_DIM   256
#define HIDDEN    512
#define B_SEEDS   1024
#define FAN1      15
#define FAN2      10

// Build a WMMA A/B operand (16 halves per lane), CDNA5 16-bit layout:
// lanes 0-15 hold K = {0..7, 16..23}, lanes 16-31 hold K = {8..15, 24..31}.
__device__ __forceinline__ v16h ld_op(const _Float16* __restrict__ p, int kb) {
  v8h lo = *(const v8h*)(p + kb);
  v8h hi = *(const v8h*)(p + 16 + kb);
  return __builtin_shufflevector(lo, hi, 0,1,2,3,4,5,6,7,8,9,10,11,12,13,14,15);
}

// acc[4] += A(16 x KD f16 in LDS, row stride astr) * W^T cols [colbase, +64).
// W row-major [out, KD] f16 in global = B-matrix columns directly.
template<int KD>
__device__ __forceinline__ void wmma_gemm(const _Float16* __restrict__ Alds, int astr,
                                          const _Float16* __restrict__ W, int colbase,
                                          int lane, v8f acc[4]) {
  const int lm = lane & 15;
  const int kb = (lane >> 4) * 8;
  const _Float16* arow = Alds + lm * astr;
#pragma unroll
  for (int k0 = 0; k0 < KD; k0 += 32) {
    v16h a = ld_op(arow + k0, kb);
#pragma unroll
    for (int t = 0; t < 4; ++t) {
      const _Float16* brow = W + (size_t)(colbase + t * 16 + lm) * KD + k0;
      v16h b = ld_op(brow, kb);
      acc[t] = __builtin_amdgcn_wmma_f32_16x16x32_f16(
          false, a, false, b, (short)0, acc[t], false, false);
    }
  }
}

// TDM: load `nrows` rows of `row_elems` f16 (global row stride
// `row_stride_elems` f16) into LDS at byte address lds_byte_addr, inserting
// 8 DWORDs (16 halves) of padding after every 1024 bytes (= one 512-half row)
// so LDS rows land at stride 528 halves. D# per CDNA5 ISA 8.3/8.4.
__device__ __forceinline__ void tdm_load_rows_f16(unsigned lds_byte_addr,
                                                  const _Float16* __restrict__ gsrc,
                                                  unsigned row_elems, unsigned nrows,
                                                  unsigned row_stride_elems) {
  unsigned long long ga = (unsigned long long)(uintptr_t)gsrc;
  u32x4 g0;
  g0[0] = 0x1u;                                           // count=1 (user D#)
  g0[1] = lds_byte_addr;                                  // lds_addr
  g0[2] = (unsigned)ga;                                   // global_addr[31:0]
  g0[3] = ((unsigned)(ga >> 32) & 0x01FFFFFFu) | 0x80000000u; // [56:32] | type=2
  u32x8 g1;
  // workgroup_mask=0 | data_size=1(2B)<<16 | pad_enable<<20 |
  // pad_interval=7 (256 DW = 1024B)<<22 | pad_amount=7 (8 DW = 32B)<<25
  g1[0] = (1u << 16) | (1u << 20) | (7u << 22) | (7u << 25);
  g1[1] = (row_elems & 0xFFFFu) << 16;                    // tensor_dim0[15:0]
  g1[2] = (row_elems >> 16) | ((nrows & 0xFFFFu) << 16);  // dim0 hi | tensor_dim1 lo
  g1[3] = (row_elems & 0xFFFFu) << 16;                    // dim1 hi(0) | tile_dim0
  g1[4] = nrows & 0xFFFFu;                                // tile_dim1 | tile_dim2=0
  g1[5] = row_stride_elems;                               // tensor_dim0_stride lo32
  g1[6] = 0u;                                             // stride0 hi | stride1 lo
  g1[7] = 0u;                                             // stride1 hi
  asm volatile("tensor_load_to_lds %0, %1" :: "s"(g0), "s"(g1) : "memory");
}

__global__ void cvt_f32_to_f16(const float* __restrict__ src,
                               _Float16* __restrict__ dst, int n) {
  int i = blockIdx.x * blockDim.x + threadIdx.x;
  if (i < n) dst[i] = (_Float16)src[i];
}

// Fused GAT layer. 16 nodes per block.
//   NEIGH_F16 = false : neighbors gathered from f32 emb via neighIdx
//   NEIGH_F16 = true  : neighbors are dense f16 rows (layer-1 output), TDM-staged
template<int FAN, int DNEIGH, bool NEIGH_F16, bool OUT_F32>
__global__ __launch_bounds__(256, 1) void gat_kernel(
    const float* __restrict__ emb,
    const int* __restrict__ selfIdx,
    const int* __restrict__ neighIdx,
    const _Float16* __restrict__ neighF16,
    const _Float16* __restrict__ Wq, const _Float16* __restrict__ Wk,
    const _Float16* __restrict__ Wv, const _Float16* __restrict__ Ws,
    _Float16* __restrict__ outH, float* __restrict__ outF) {
  constexpr int SSTR = EMB_DIM + 16;   // LDS row stride (halves), padded
  constexpr int NSTR = DNEIGH + 16;
  __shared__ _Float16 sSelf[16 * SSTR];
  __shared__ _Float16 sNbr[16 * NSTR];
  __shared__ float    sQ[8 * 8 * 32 * 4];  // [wave][i][lane][t] : 32 KB

  const int tid = threadIdx.x;
  const int lane = tid & 31;
  const int wave = tid >> 5;
  const int nodeBase = blockIdx.x * 16;
  const int colBase = wave * 64;   // wave = head (HIDDEN/HEADS = 64)

  // --- stage self rows: gather f32 emb -> f16 LDS (16 rows x 256) ---------
  {
    const int r = tid >> 4, seg = tid & 15;
    const int row = selfIdx[nodeBase + r];
    const float4* src = (const float4*)(emb + (size_t)row * EMB_DIM + seg * 16);
    _Float16* dst = sSelf + r * SSTR + seg * 16;
    float4 a0 = src[0], a1 = src[1], a2 = src[2], a3 = src[3];
    v8h h0 = {(_Float16)a0.x, (_Float16)a0.y, (_Float16)a0.z, (_Float16)a0.w,
              (_Float16)a1.x, (_Float16)a1.y, (_Float16)a1.z, (_Float16)a1.w};
    v8h h1 = {(_Float16)a2.x, (_Float16)a2.y, (_Float16)a2.z, (_Float16)a2.w,
              (_Float16)a3.x, (_Float16)a3.y, (_Float16)a3.z, (_Float16)a3.w};
    *(v8h*)(dst) = h0;
    *(v8h*)(dst + 8) = h1;
  }
  __syncthreads();

  const v8f vzero = {0.f, 0.f, 0.f, 0.f, 0.f, 0.f, 0.f, 0.f};

  // Q = x_self @ Wq^T for this wave's head, then parked in LDS (per-wave
  // private region) to free 32 VGPRs for the fanout loop.
  {
    v8f qacc[4];
#pragma unroll
    for (int t = 0; t < 4; ++t) qacc[t] = vzero;
    wmma_gemm<EMB_DIM>(sSelf, SSTR, Wq, colBase, lane, qacc);
#pragma unroll
    for (int i = 0; i < 8; ++i)
#pragma unroll
      for (int t = 0; t < 4; ++t)
        sQ[wave * 1024 + i * 128 + lane * 4 + t] = qacc[t][i];
  }

  v8f agg[4], kv[4];
#pragma unroll
  for (int t = 0; t < 4; ++t) agg[t] = vzero;

  // Online-softmax state, per row (C layout: vgpr i <-> rows {i, i+8}).
  float mrow[8], srow[8];
#pragma unroll
  for (int i = 0; i < 8; ++i) { mrow[i] = -3.0e38f; srow[i] = 0.f; }

  for (int f = 0; f < FAN; ++f) {
    __syncthreads();  // all waves done reading sNbr from previous iteration
    if constexpr (NEIGH_F16) {
      // One TDM descriptor moves the whole 16 x DNEIGH f16 tile, with D#
      // padding producing the bank-conflict-free NSTR row stride in LDS.
      if (wave == 0) {
        const _Float16* src = neighF16 + ((size_t)nodeBase * FAN + f) * DNEIGH;
        tdm_load_rows_f16((unsigned)(uintptr_t)sNbr, src,
                          (unsigned)DNEIGH, 16u, (unsigned)(FAN * DNEIGH));
        __builtin_amdgcn_s_wait_tensorcnt(0);
      }
    } else {
      const int r = tid >> 4, seg = tid & 15;
      const int row = neighIdx[(nodeBase + r) * FAN + f];
      const float4* src = (const float4*)(emb + (size_t)row * EMB_DIM + seg * 16);
      _Float16* dst = sNbr + r * NSTR + seg * 16;
      float4 a0 = src[0], a1 = src[1], a2 = src[2], a3 = src[3];
      v8h h0 = {(_Float16)a0.x, (_Float16)a0.y, (_Float16)a0.z, (_Float16)a0.w,
                (_Float16)a1.x, (_Float16)a1.y, (_Float16)a1.z, (_Float16)a1.w};
      v8h h1 = {(_Float16)a2.x, (_Float16)a2.y, (_Float16)a2.z, (_Float16)a2.w,
                (_Float16)a3.x, (_Float16)a3.y, (_Float16)a3.z, (_Float16)a3.w};
      *(v8h*)(dst) = h0;
      *(v8h*)(dst + 8) = h1;
    }
    __syncthreads();

    // K = x_nbr @ Wk^T for this head
#pragma unroll
    for (int t = 0; t < 4; ++t) kv[t] = vzero;
    wmma_gemm<DNEIGH>(sNbr, NSTR, Wk, colBase, lane, kv);

    // score[row] = (K . Q): per-VGPR product + 16-lane xor butterfly leaves
    // the score broadcast in exactly the C-matrix lane layout.
    float sv[8];
#pragma unroll
    for (int i = 0; i < 8; ++i) {
      const float4 qv = *(const float4*)(sQ + wave * 1024 + i * 128 + lane * 4);
      float p = kv[0][i] * qv.x + kv[1][i] * qv.y + kv[2][i] * qv.z + kv[3][i] * qv.w;
      p += __shfl_xor(p, 1, 32);
      p += __shfl_xor(p, 2, 32);
      p += __shfl_xor(p, 4, 32);
      p += __shfl_xor(p, 8, 32);
      sv[i] = p * 0.125f;  // hd^-0.5, hd = 64
    }

    // V = x_nbr @ Wv^T (reuse kv accumulators)
#pragma unroll
    for (int t = 0; t < 4; ++t) kv[t] = vzero;
    wmma_gemm<DNEIGH>(sNbr, NSTR, Wv, colBase, lane, kv);

    // Online softmax update: agg = agg*corr + exp(s-m)*V
#pragma unroll
    for (int i = 0; i < 8; ++i) {
      float nm = fmaxf(mrow[i], sv[i]);
      float c = __expf(mrow[i] - nm);
      float pe = __expf(sv[i] - nm);
      srow[i] = srow[i] * c + pe;
      mrow[i] = nm;
#pragma unroll
      for (int t = 0; t < 4; ++t) agg[t][i] = agg[t][i] * c + pe * kv[t][i];
    }
  }

  // self-transform + combine: out = relu(x_self @ Ws^T + agg/sum)
#pragma unroll
  for (int t = 0; t < 4; ++t) kv[t] = vzero;
  wmma_gemm<EMB_DIM>(sSelf, SSTR, Ws, colBase, lane, kv);

  const int lm = lane & 15;
  const int rAdd = (lane >> 4) * 8;
#pragma unroll
  for (int i = 0; i < 8; ++i) {
    const float inv = 1.0f / srow[i];
    const int node = nodeBase + i + rAdd;
#pragma unroll
    for (int t = 0; t < 4; ++t) {
      float o = fmaxf(kv[t][i] + agg[t][i] * inv, 0.f);
      const int col = colBase + t * 16 + lm;
      if (OUT_F32) outF[(size_t)node * HIDDEN + col] = o;
      else         outH[(size_t)node * HIDDEN + col] = (_Float16)o;
    }
  }
}

// ---------------------------------------------------------------------------
// Workspace layout (in halves)
// ---------------------------------------------------------------------------
#define OFF_WQ1 0u
#define OFF_WK1 131072u
#define OFF_WV1 262144u
#define OFF_WS1 393216u
#define OFF_WQ2 524288u
#define OFF_WK2 655360u
#define OFF_WV2 917504u
#define OFF_WS2 1179648u
#define OFF_H1  1310720u    // 15360 * 512 halves; total ws ~18.4 MB

extern "C" void kernel_launch(void* const* d_in, const int* in_sizes, int n_in,
                              void* d_out, int out_size, void* d_ws, size_t ws_size,
                              hipStream_t stream) {
  const int* seeds = (const int*)d_in[0];
  const int* nbr1  = (const int*)d_in[1];
  const int* nbr2  = (const int*)d_in[2];
  const float* emb = (const float*)d_in[3];
  _Float16* w = (_Float16*)d_ws;

  // Convert weights f32 -> f16 into workspace (redone each call: deterministic).
  const int widx[8] = {4, 5, 6, 7, 8, 9, 10, 11};
  const unsigned woff[8] = {OFF_WQ1, OFF_WK1, OFF_WV1, OFF_WS1,
                            OFF_WQ2, OFF_WK2, OFF_WV2, OFF_WS2};
  const int wn[8] = {HIDDEN * EMB_DIM, HIDDEN * EMB_DIM, HIDDEN * EMB_DIM,
                     HIDDEN * EMB_DIM, HIDDEN * EMB_DIM, HIDDEN * HIDDEN,
                     HIDDEN * HIDDEN, HIDDEN * EMB_DIM};
  for (int i = 0; i < 8; ++i) {
    int n = wn[i];
    cvt_f32_to_f16<<<(n + 255) / 256, 256, 0, stream>>>(
        (const float*)d_in[widx[i]], w + woff[i], n);
  }

  _Float16* H1 = w + OFF_H1;

  // Layer 1: 15360 nodes, fanout 10, neighbors gathered from f32 emb.
  gat_kernel<FAN2, EMB_DIM, false, false><<<(B_SEEDS * FAN1) / 16, 256, 0, stream>>>(
      emb, nbr1, nbr2, (const _Float16*)nullptr,
      w + OFF_WQ1, w + OFF_WK1, w + OFF_WV1, w + OFF_WS1,
      H1, (float*)nullptr);

  // Layer 2: 1024 seed nodes, fanout 15, neighbors are f16 H1 rows via TDM.
  gat_kernel<FAN1, HIDDEN, true, true><<<B_SEEDS / 16, 256, 0, stream>>>(
      emb, seeds, (const int*)nullptr, H1,
      w + OFF_WQ2, w + OFF_WK2, w + OFF_WV2, w + OFF_WS2,
      (_Float16*)nullptr, (float*)d_out);
}
